// BCNet_77275051590189
// MI455X (gfx1250) — matrix operand order repaired and checked by reading, not compiled, not run on hardware
//
#include <hip/hip_runtime.h>

typedef float  v4f  __attribute__((ext_vector_type(4)));
typedef float  v8f  __attribute__((ext_vector_type(8)));
typedef __bf16 v4bf __attribute__((ext_vector_type(4)));
typedef __bf16 v8bf __attribute__((ext_vector_type(8)));
typedef __bf16 v16bf __attribute__((ext_vector_type(16)));
typedef int    v4i_gcc __attribute__((vector_size(16)));   // matches builtin param type

#define WMMA_BF16(A, B, C) \
    __builtin_amdgcn_wmma_f32_16x16x32_bf16(false, (A), false, (B), (short)0, (C), false, false)

__device__ __forceinline__ v16bf cat8(v8bf a, v8bf b) {
    return __builtin_shufflevector(a, b, 0,1,2,3,4,5,6,7,8,9,10,11,12,13,14,15);
}

// ---------------------------------------------------------------------------
// Async global->LDS copy (CDNA5 GLOBAL_LOAD_ASYNC_TO_LDS_B128, ASYNCcnt).
// Toolchain-probed signature: (AS1 v4i*, AS3 v4i*, imm offset, imm cpol).
// Guarded: falls back to VGPR staging if the builtin is not declared.
// ---------------------------------------------------------------------------
#define AS_G __attribute__((address_space(1)))
#define AS_L __attribute__((address_space(3)))

#if defined(__AMDGCN__) && __has_builtin(__builtin_amdgcn_global_load_async_to_lds_b128)
#define HAVE_ASYNC_LDS 1
#else
#define HAVE_ASYNC_LDS 0
#endif

__device__ __forceinline__ void cp16_g2l(const __bf16* g, __bf16* l) {
#if HAVE_ASYNC_LDS
    // generic->AS casts via integer: LDS generic addr low 32 bits == LDS offset
    __builtin_amdgcn_global_load_async_to_lds_b128(
        (AS_G v4i_gcc*)(unsigned long long)g,
        (AS_L v4i_gcc*)(unsigned int)(unsigned long long)l, 0, 0);
#else
    *(v8bf*)l = *(const v8bf*)g;
#endif
}

__device__ __forceinline__ void wait_cp_g2l() {
#if HAVE_ASYNC_LDS
#if __has_builtin(__builtin_amdgcn_s_wait_asynccnt)
    __builtin_amdgcn_s_wait_asynccnt(0);
#else
    asm volatile("s_wait_asynccnt 0x0" ::: "memory");
#endif
#endif
}

// ---------------------------------------------------------------------------
// f32 -> bf16 bulk conversion (one pass; GEMM hot loops then run pure bf16)
// ---------------------------------------------------------------------------
__global__ __launch_bounds__(256)
void f32_to_bf16_kernel(const float* __restrict__ s, __bf16* __restrict__ d, int n8) {
    const int i = blockIdx.x * 256 + threadIdx.x;
    if (i < n8) {
        const float* p = s + (size_t)i * 8;
        v4f f0 = *(const v4f*)p;
        v4f f1 = *(const v4f*)(p + 4);
        v8f f = __builtin_shufflevector(f0, f1, 0,1,2,3,4,5,6,7);
        *(v8bf*)(d + (size_t)i * 8) = __builtin_convertvector(f, v8bf);
    }
}

// ---------------------------------------------------------------------------
// Kernel 1: q_ = ReLU(q @ Wq^T + bq); qw[b,h] = sum_q q_[b,q,h]*wh[q]
// Grid: (B, H/64). Block: 128 threads = 4 waves; wave w owns N-tile w.
// ---------------------------------------------------------------------------
__global__ __launch_bounds__(128)
void qnet_qw_kernel(const float* __restrict__ q, const float* __restrict__ Wq,
                    const float* __restrict__ bq, const float* __restrict__ wh,
                    float* __restrict__ qw)
{
    constexpr int Qn = 14, QD = 1024, H = 2048, BK = 32, LDT = BK + 8;
    __shared__ __bf16 As[16 * LDT];
    __shared__ __bf16 Bs[64 * LDT];

    const int tid  = threadIdx.x;
    const int wave = tid >> 5;
    const int lane = tid & 31;
    const int l16  = lane & 15;
    const int hi   = lane >> 4;

    const int b  = blockIdx.x;
    const int h0 = blockIdx.y * 64;

    v8f acc = {};

    const int ar = tid >> 3;
    const int ak = (tid & 7) << 2;
    const int arc = ar < Qn ? ar : Qn - 1;
    const float* Ap = q + ((size_t)b * Qn + arc) * QD + ak;
    const int br = tid >> 1;
    const int bk = (tid & 1) << 4;
    const float* Wp = Wq + (size_t)(h0 + br) * QD + bk;

    for (int kb = 0; kb < QD / BK; ++kb) {
        const int k0 = kb * BK;
        v4f fa = *(const v4f*)(Ap + k0);
        *(v4bf*)&As[ar * LDT + ak] = __builtin_convertvector(fa, v4bf);
        v4f f0 = *(const v4f*)(Wp + k0);
        v4f f1 = *(const v4f*)(Wp + k0 + 4);
        v4f f2 = *(const v4f*)(Wp + k0 + 8);
        v4f f3 = *(const v4f*)(Wp + k0 + 12);
        v8f g0 = __builtin_shufflevector(f0, f1, 0,1,2,3,4,5,6,7);
        v8f g1 = __builtin_shufflevector(f2, f3, 0,1,2,3,4,5,6,7);
        *(v8bf*)&Bs[br * LDT + bk]     = __builtin_convertvector(g0, v8bf);
        *(v8bf*)&Bs[br * LDT + bk + 8] = __builtin_convertvector(g1, v8bf);
        __syncthreads();

        const int ab = l16 * LDT + hi * 8;
        v16bf af = cat8(*(const v8bf*)&As[ab], *(const v8bf*)&As[ab + 16]);
        const int bb = (wave * 16 + l16) * LDT + hi * 8;
        v16bf bf = cat8(*(const v8bf*)&Bs[bb], *(const v8bf*)&Bs[bb + 16]);
        acc = WMMA_BF16(af, bf, acc);
        __syncthreads();
    }

    const int col = h0 + wave * 16 + l16;
    const float bqc = bq[col];
    float partial = 0.0f;
    #pragma unroll
    for (int i = 0; i < 8; ++i) {
        const int row = i + hi * 8;
        float x = fmaxf(acc[i] + bqc, 0.0f);
        float w = (row < Qn) ? wh[row] : 0.0f;
        partial += w * x;
    }
    partial += __shfl_xor(partial, 16, 32);
    if (hi == 0) qw[(size_t)b * H + col] = partial;
}

// ---------------------------------------------------------------------------
// Kernels 2/3: pure-bf16 GEMM  C = epi(A[MxK] @ W[NxK]^T).
// 128x128 block, BK=64, 256 thr = 8 waves; wave = 32(M) x 64(N) patch =
// 2x4 accumulators -> 16 v_wmma per k-block vs 24 ds_load_b128.
// Double-buffered LDS, async global->LDS staging overlapped with WMMA.
// EPI==0: bf16( ReLU(x+bias[col]) * qw[row/36,col] + bh )  (logits, bf16 out)
// EPI==1: f32 ( x + bias[col] )                            (final out)
// ---------------------------------------------------------------------------
template<int EPI>
__global__ __launch_bounds__(256)
void gemm_bf16_128x128(const __bf16* __restrict__ A, const __bf16* __restrict__ W,
                       const float* __restrict__ bias, const float* __restrict__ qw,
                       const float* __restrict__ bh_ptr, void* __restrict__ out,
                       int M, int N, int K)
{
    constexpr int BM = 128, BN = 128, BK = 64, LDT = BK + 8;   // 72 bf16/row
    __shared__ __bf16 As[2][BM * LDT];
    __shared__ __bf16 Bs[2][BN * LDT];

    const int tid  = threadIdx.x;
    const int wave = tid >> 5;
    const int lane = tid & 31;
    const int l16  = lane & 15;
    const int hi   = lane >> 4;

    const int m0 = blockIdx.y * BM;
    const int n0 = blockIdx.x * BN;
    const int wm = wave & 3;    // m-pair: tiles {2wm, 2wm+1}
    const int wn = wave >> 2;   // n-quad: tiles {4wn .. 4wn+3}

    v8f acc[2][4] = {};

    // Staging map: 8 chunks of 16B per 128B row; thread t copies 4 consecutive
    // chunks (64B contiguous global) for A and for W.
    const int srow = tid >> 1;            // 0..127
    const int sch  = (tid & 1) << 2;      // 0 or 4 (chunk base)
    const __bf16* Ag = A + (size_t)(m0 + srow) * K + sch * 8;
    const __bf16* Wg = W + (size_t)(n0 + srow) * K + sch * 8;

    const int nk = K / BK;

    // stage tile kb into buffer buf
    auto stage = [&](int kb, int buf) {
        const __bf16* ga = Ag + (size_t)kb * BK;
        const __bf16* gw = Wg + (size_t)kb * BK;
        __bf16* la = &As[buf][srow * LDT + sch * 8];
        __bf16* lb = &Bs[buf][srow * LDT + sch * 8];
        #pragma unroll
        for (int j = 0; j < 4; ++j) {
            cp16_g2l(ga + j * 8, la + j * 8);
            cp16_g2l(gw + j * 8, lb + j * 8);
        }
    };

    stage(0, 0);
    wait_cp_g2l();
    __syncthreads();

    for (int kb = 0; kb < nk; ++kb) {
        const int cur = kb & 1;
        if (kb + 1 < nk) stage(kb + 1, cur ^ 1);

        const __bf16* sA = As[cur];
        const __bf16* sB = Bs[cur];
        #pragma unroll
        for (int ks = 0; ks < BK; ks += 32) {
            v16bf afr[2], bfr[4];
            #pragma unroll
            for (int mi = 0; mi < 2; ++mi) {
                const int ab = ((wm * 2 + mi) * 16 + l16) * LDT + ks + hi * 8;
                afr[mi] = cat8(*(const v8bf*)&sA[ab], *(const v8bf*)&sA[ab + 16]);
            }
            #pragma unroll
            for (int ni = 0; ni < 4; ++ni) {
                const int bb = ((wn * 4 + ni) * 16 + l16) * LDT + ks + hi * 8;
                bfr[ni] = cat8(*(const v8bf*)&sB[bb], *(const v8bf*)&sB[bb + 16]);
            }
            #pragma unroll
            for (int mi = 0; mi < 2; ++mi)
                #pragma unroll
                for (int ni = 0; ni < 4; ++ni)
                    acc[mi][ni] = WMMA_BF16(afr[mi], bfr[ni], acc[mi][ni]);
        }

        if (kb + 1 < nk) wait_cp_g2l();
        __syncthreads();
    }

    const float bh = (EPI == 0) ? *bh_ptr : 0.0f;
    #pragma unroll
    for (int mi = 0; mi < 2; ++mi) {
        #pragma unroll
        for (int ni = 0; ni < 4; ++ni) {
            v8f a = acc[mi][ni];
            const int col = n0 + (wn * 4 + ni) * 16 + l16;
            const float bc = bias[col];
            #pragma unroll
            for (int i = 0; i < 8; ++i) {
                const int row = m0 + (wm * 2 + mi) * 16 + hi * 8 + i;
                float x = a[i];
                if constexpr (EPI == 0) {
                    x = fmaxf(x + bc, 0.0f);
                    x = x * qw[(size_t)(row / 36) * N + col] + bh;   // NO = 36
                    ((__bf16*)out)[(size_t)row * N + col] = (__bf16)x;
                } else {
                    ((float*)out)[(size_t)row * N + col] = x + bc;
                }
            }
        }
    }
}

extern "C" void kernel_launch(void* const* d_in, const int* in_sizes, int n_in,
                              void* d_out, int out_size, void* d_ws, size_t ws_size,
                              hipStream_t stream)
{
    (void)in_sizes; (void)n_in; (void)out_size; (void)ws_size;
    const float* v  = (const float*)d_in[0];
    const float* q  = (const float*)d_in[1];
    const float* Wv = (const float*)d_in[2];
    const float* bv = (const float*)d_in[3];
    const float* Wq = (const float*)d_in[4];
    const float* bq = (const float*)d_in[5];
    const float* wh = (const float*)d_in[6];
    const float* bh = (const float*)d_in[7];
    const float* W2 = (const float*)d_in[8];
    const float* b2 = (const float*)d_in[9];
    float* out = (float*)d_out;

    constexpr int B = 128, NO = 36, VD = 2048, H = 2048;
    constexpr int M = B * NO;                               // 4608

    // workspace layout
    float*  qw    = (float*)d_ws;                           // B*H f32     1 MB
    __bf16* vbf   = (__bf16*)(qw + (size_t)B * H);          // M*VD bf16  18.9 MB
    __bf16* Wvbf  = vbf  + (size_t)M * VD;                  // H*VD bf16   8.4 MB
    __bf16* W2bf  = Wvbf + (size_t)H * VD;                  // VD*H bf16   8.4 MB
    __bf16* logbf = W2bf + (size_t)VD * H;                  // M*H  bf16  18.9 MB

    // 0) one-pass f32->bf16 of the GEMM operands (L2-resident afterwards)
    f32_to_bf16_kernel<<<(M * VD / 8 + 255) / 256, 256, 0, stream>>>(v,  vbf,  M * VD / 8);
    f32_to_bf16_kernel<<<(H * VD / 8 + 255) / 256, 256, 0, stream>>>(Wv, Wvbf, H * VD / 8);
    f32_to_bf16_kernel<<<(VD * H / 8 + 255) / 256, 256, 0, stream>>>(W2, W2bf, VD * H / 8);

    // 1) qw[b,h] (fused q_net + wh reduction)
    qnet_qw_kernel<<<dim3(B, H / 64), 128, 0, stream>>>(q, Wq, bq, wh, qw);

    // 2) logits(bf16) = ReLU(v @ Wv^T + bv) * qw + bh
    gemm_bf16_128x128<0><<<dim3(H / 128, M / 128), 256, 0, stream>>>(
        vbf, Wvbf, bv, qw, bh, (void*)logbf, M, H, VD);

    // 3) out(f32) = logits @ W2^T + b2
    gemm_bf16_128x128<1><<<dim3(VD / 128, M / 128), 256, 0, stream>>>(
        logbf, W2bf, b2, nullptr, nullptr, (void*)out, M, VD, H);
}
